// GridTorch_57853209477120
// MI455X (gfx1250) — matrix operand order repaired and checked by reading, not compiled, not run on hardware
//
#include <hip/hip_runtime.h>

#define T_STEPS 100
#define BATCH   1024
#define NHID    128
#define NBOT    256
#define NPCELL  256
#define NHDIR   12
#define NICOND  268

typedef __attribute__((ext_vector_type(16))) _Float16 v16h;
typedef __attribute__((ext_vector_type(8)))  float    v8f;

union H16 { unsigned u[8]; v16h h; };

static __device__ __forceinline__ v8f wmma16(v16h a, v16h b, v8f c) {
  // D = A(16x32 f16) x B(32x16 f16) + C(16x16 f32)
  return __builtin_amdgcn_wmma_f32_16x16x32_f16(false, a, false, b, (short)0, c,
                                                false, false);
}

static __device__ __forceinline__ unsigned pk2h(float a, float b) {
  _Float16 ha = (_Float16)a, hb = (_Float16)b;
  unsigned short ua, ub;
  __builtin_memcpy(&ua, &ha, 2);
  __builtin_memcpy(&ub, &hb, 2);
  return (unsigned)ua | ((unsigned)ub << 16);
}

static __device__ __forceinline__ float sigm(float x) {
  return 1.0f / (1.0f + __expf(-x));
}

// A fragment: row-major f16 matrix in LDS viewed as dwords (ld2 dwords/row).
// Layout per ISA 7.12.2: lanes 0-15 rows M=0..15 K-low half, lanes 16-31 K-high half,
// vgpr v<4 -> K pairs 2v, v>=4 -> 16+2(v-4).
static __device__ __forceinline__ v16h ldA(const unsigned* a, int ld2, int m0,
                                           int k0, int lane) {
  int m = m0 + (lane & 15);
  int base = m * ld2 + (k0 >> 1) + ((lane >> 4) << 2);
  H16 f;
#pragma unroll
  for (int v = 0; v < 8; ++v) f.u[v] = a[base + (v < 4 ? v : v + 4)];
  return f.h;
}

// B fragment: weights K-pair-packed (dword row = k/2, ldn dwords/row).
// Lanes 0-15: K=0..7,16..23, lanes 16-31: K=8..15,24..31; n = n0 + lane&15.
static __device__ __forceinline__ v16h ldB(const unsigned* w, int ldn, int k0,
                                           int n0, int lane) {
  int n = n0 + (lane & 15);
  int kr = (k0 >> 1) + ((lane >> 4) << 2);
  H16 f;
#pragma unroll
  for (int v = 0; v < 8; ++v) f.u[v] = w[(kr + (v < 4 ? v : v + 4)) * ldn + n];
  return f.h;
}

// ---------- pre-kernel: pack pc_w fp32[256][256] -> K-pair f16 dwords ----------
__global__ void pack_pc_kernel(const float* __restrict__ pc_w,
                               unsigned* __restrict__ pcwp) {
  int idx = blockIdx.x * 256 + threadIdx.x;  // 32768
  int row = idx >> 8, n = idx & 255, k = row << 1;
  pcwp[idx] = pk2h(pc_w[k * 256 + n], pc_w[(k + 1) * 256 + n]);
}

// ---------- pre-kernel: h0 = init@state_w + b, c0 = init@cell_w + b ----------
__global__ void init_hc_kernel(const float* __restrict__ init,
                               const float* __restrict__ sw,
                               const float* __restrict__ sb,
                               const float* __restrict__ cw,
                               const float* __restrict__ cb,
                               float* __restrict__ h0, float* __restrict__ c0) {
  int idx = blockIdx.x * 256 + threadIdx.x;  // 1024*128
  int b = idx >> 7, n = idx & 127;
  const float* ip = init + b * NICOND;
  float ah = sb[n], ac = cb[n];
  for (int k = 0; k < NICOND; ++k) {
    float xv = ip[k];
    ah = fmaf(xv, sw[k * 128 + n], ah);
    ac = fmaf(xv, cw[k * 128 + n], ac);
  }
  h0[idx] = ah;
  c0[idx] = ac;
}

// ---------- persistent LSTM kernel: one WG per 32-row batch tile ----------
__launch_bounds__(256, 1)
__global__ void lstm_kernel(const float* __restrict__ x,
                            const float* __restrict__ rnn_w,
                            const float* __restrict__ rnn_b,
                            const float* __restrict__ bneck_w,
                            const float* __restrict__ pc_b,
                            const float* __restrict__ hd_w,
                            const float* __restrict__ hd_b,
                            const unsigned* __restrict__ pcwp,
                            const float* __restrict__ h0,
                            const float* __restrict__ c0,
                            float* __restrict__ out) {
  extern __shared__ char smem[];
  unsigned* whp    = (unsigned*)smem;       // [64][512]  rnn_w rows 3..130, packed (128 KB)
  unsigned* wbnp   = whp + 32768;           // [64][256]  bneck_w packed      (64 KB)
  unsigned* whdp   = wbnp + 16384;          // [128][16]  hd_w packed, N pad  (8 KB)
  unsigned* hbufU  = whdp + 2048;           // [32][128]  h_t f16             (8 KB)
  unsigned* bnbufU = hbufU + 2048;          // [32][256]  bn f16              (16 KB)
  float* wx  = (float*)(bnbufU + 4096);     // [3][512]   rnn_w rows 0..2 fp32
  float* rb  = wx + 1536;                   // [512] rnn_b
  float* pcb = rb + 512;                    // [256] pc_b
  float* hdb = pcb + 256;                   // [16]  hd_b padded
  float* xb  = hdb + 16;                    // [32][3] x_t tile
  _Float16* hbufH  = (_Float16*)hbufU;
  _Float16* bnbufH = (_Float16*)bnbufU;

  const int tid = threadIdx.x;
  const int wave = tid >> 5, lane = tid & 31;
  const int col = lane & 15, rowb = (lane >> 4) << 3;  // D-layout: vgpr v -> row rowb+v
  const int batch0 = blockIdx.x * 32;

  float* outHD = out;
  float* outPC = outHD + (size_t)T_STEPS * BATCH * NHDIR;
  float* outBN = outPC + (size_t)T_STEPS * BATCH * NPCELL;
  float* outHS = outBN + (size_t)T_STEPS * BATCH * NBOT;
  float* outCS = outHS + (size_t)T_STEPS * BATCH * NHID;

  // ---- stage weights into LDS (once) ----
  for (int i = tid; i < 32768; i += 256) {
    int r = i >> 9, n = i & 511, k = (r << 1) + 3;  // skip the 3 x-input rows
    whp[i] = pk2h(rnn_w[k * 512 + n], rnn_w[(k + 1) * 512 + n]);
  }
  for (int i = tid; i < 16384; i += 256) {
    int r = i >> 8, n = i & 255, k = r << 1;
    wbnp[i] = pk2h(bneck_w[k * 256 + n], bneck_w[(k + 1) * 256 + n]);
  }
  for (int i = tid; i < 2048; i += 256) {
    int r = i >> 4, n = i & 15, k = r << 1;
    whdp[i] = (n < NHDIR) ? pk2h(hd_w[k * NHDIR + n], hd_w[(k + 1) * NHDIR + n]) : 0u;
  }
  for (int i = tid; i < 1536; i += 256) wx[i] = rnn_w[i];
  for (int i = tid; i < 512; i += 256) rb[i] = rnn_b[i];
  if (tid < 256) pcb[tid] = pc_b[tid];
  if (tid < 16) hdb[tid] = (tid < NHDIR) ? hd_b[tid] : 0.0f;
  for (int i = tid; i < 4096; i += 256) {
    int r = i >> 7, n = i & 127;
    hbufH[i] = (_Float16)h0[(batch0 + r) * 128 + n];
  }

  // persistent cell state: wave w owns hidden cols [16w,16w+16), m-tiles 0/1
  v8f cacc[2];
#pragma unroll
  for (int mt = 0; mt < 2; ++mt)
#pragma unroll
    for (int v = 0; v < 8; ++v)
      cacc[mt][v] = c0[(batch0 + mt * 16 + rowb + v) * 128 + wave * 16 + col];

  __syncthreads();

  for (int t = 0; t < T_STEPS; ++t) {
    if (tid < 96) xb[tid] = x[(size_t)(t * BATCH + batch0) * 3 + tid];
    __syncthreads();  // xb ready; also fences prev-step bnbuf readers

    // ---- gates = [x_t, h] @ rnn_w + rnn_b  (wave w -> gate cols 16w..16w+15) ----
    v8f g[2][4];
#pragma unroll
    for (int mt = 0; mt < 2; ++mt) {
      float xr[8][3];
#pragma unroll
      for (int v = 0; v < 8; ++v) {
        int r = mt * 16 + rowb + v;
        xr[v][0] = xb[r * 3 + 0];
        xr[v][1] = xb[r * 3 + 1];
        xr[v][2] = xb[r * 3 + 2];
      }
#pragma unroll
      for (int gg = 0; gg < 4; ++gg) {  // i, j, f, o
        int n = gg * 128 + wave * 16 + col;
        float b = rb[n], w0 = wx[n], w1 = wx[512 + n], w2 = wx[1024 + n];
        v8f a;
#pragma unroll
        for (int v = 0; v < 8; ++v)
          a[v] = b + xr[v][0] * w0 + xr[v][1] * w1 + xr[v][2] * w2;
        g[mt][gg] = a;
      }
#pragma unroll
      for (int kc = 0; kc < 4; ++kc) {
        v16h af = ldA(hbufU, 64, mt * 16, kc * 32, lane);
#pragma unroll
        for (int gg = 0; gg < 4; ++gg) {
          v16h bf = ldB(whp, 512, kc * 32, gg * 128 + wave * 16, lane);
          g[mt][gg] = wmma16(af, bf, g[mt][gg]);
        }
      }
    }
    __syncthreads();  // all h_{t-1} reads done

    // ---- elementwise LSTM; c stays in registers, h -> LDS f16 + global f32 ----
#pragma unroll
    for (int mt = 0; mt < 2; ++mt) {
#pragma unroll
      for (int v = 0; v < 8; ++v) {
        float iv = g[mt][0][v], jv = g[mt][1][v], fv = g[mt][2][v], ov = g[mt][3][v];
        float cn = cacc[mt][v] * sigm(fv + 1.0f) + sigm(iv) * tanhf(jv);
        float hn = tanhf(cn) * sigm(ov);
        cacc[mt][v] = cn;
        int r = mt * 16 + rowb + v, hc = wave * 16 + col;
        hbufH[r * 128 + hc] = (_Float16)hn;
        size_t gb = (size_t)(t * BATCH + batch0 + r) * NHID + hc;
        outHS[gb] = hn;
        outCS[gb] = cn;
      }
    }
    __syncthreads();  // h_t ready

    // ---- bn = h_t @ bneck_w (no bias); wave w -> n-tiles {2w, 2w+1} ----
    v8f bacc[2][2];
#pragma unroll
    for (int mt = 0; mt < 2; ++mt) {
#pragma unroll
      for (int ns = 0; ns < 2; ++ns) { v8f z = {}; bacc[mt][ns] = z; }
#pragma unroll
      for (int kc = 0; kc < 4; ++kc) {
        v16h af = ldA(hbufU, 64, mt * 16, kc * 32, lane);
#pragma unroll
        for (int ns = 0; ns < 2; ++ns) {
          v16h bf = ldB(wbnp, 256, kc * 32, (wave * 2 + ns) * 16, lane);
          bacc[mt][ns] = wmma16(af, bf, bacc[mt][ns]);
        }
      }
#pragma unroll
      for (int ns = 0; ns < 2; ++ns)
#pragma unroll
        for (int v = 0; v < 8; ++v) {
          float bv = bacc[mt][ns][v];
          int r = mt * 16 + rowb + v, n = (wave * 2 + ns) * 16 + col;
          bnbufH[r * 256 + n] = (_Float16)bv;
          outBN[(size_t)(t * BATCH + batch0 + r) * NBOT + n] = bv;
        }
    }
    __syncthreads();  // bn ready

    // ---- pc = bn @ pc_w + pc_b (B streamed from L2-resident packed copy) ----
    v8f pacc[2][2];
#pragma unroll
    for (int mt = 0; mt < 2; ++mt) {
#pragma unroll
      for (int ns = 0; ns < 2; ++ns) {
        float b = pcb[(wave * 2 + ns) * 16 + col];
        v8f a;
#pragma unroll
        for (int v = 0; v < 8; ++v) a[v] = b;
        pacc[mt][ns] = a;
      }
#pragma unroll
      for (int kc = 0; kc < 8; ++kc) {
        v16h af = ldA(bnbufU, 128, mt * 16, kc * 32, lane);
#pragma unroll
        for (int ns = 0; ns < 2; ++ns) {
          v16h bf = ldB(pcwp, 256, kc * 32, (wave * 2 + ns) * 16, lane);
          pacc[mt][ns] = wmma16(af, bf, pacc[mt][ns]);
        }
      }
#pragma unroll
      for (int ns = 0; ns < 2; ++ns)
#pragma unroll
        for (int v = 0; v < 8; ++v) {
          int r = mt * 16 + rowb + v, n = (wave * 2 + ns) * 16 + col;
          outPC[(size_t)(t * BATCH + batch0 + r) * NPCELL + n] = pacc[mt][ns][v];
        }
    }

    // ---- hd = bn @ hd_w + hd_b (N padded 12->16; waves 0/1, wave-uniform) ----
    if (wave < 2) {
      int mt = wave;
      float b = hdb[col];
      v8f hacc;
#pragma unroll
      for (int v = 0; v < 8; ++v) hacc[v] = b;
#pragma unroll
      for (int kc = 0; kc < 8; ++kc) {
        v16h af = ldA(bnbufU, 128, mt * 16, kc * 32, lane);
        v16h bf = ldB(whdp, 16, kc * 32, 0, lane);
        hacc = wmma16(af, bf, hacc);
      }
      if (col < NHDIR) {
#pragma unroll
        for (int v = 0; v < 8; ++v) {
          int r = mt * 16 + rowb + v;
          outHD[(size_t)(t * BATCH + batch0 + r) * NHDIR + col] = hacc[v];
        }
      }
    }
    // top-of-loop barrier fences bnbuf readers before next overwrite
  }
}

extern "C" void kernel_launch(void* const* d_in, const int* in_sizes, int n_in,
                              void* d_out, int out_size, void* d_ws, size_t ws_size,
                              hipStream_t stream) {
  (void)in_sizes; (void)n_in; (void)out_size; (void)ws_size;
  const float* x       = (const float*)d_in[0];
  const float* init    = (const float*)d_in[1];
  const float* rnn_w   = (const float*)d_in[2];
  const float* rnn_b   = (const float*)d_in[3];
  const float* state_w = (const float*)d_in[4];
  const float* state_b = (const float*)d_in[5];
  const float* cell_w  = (const float*)d_in[6];
  const float* cell_b  = (const float*)d_in[7];
  const float* bneck_w = (const float*)d_in[8];
  const float* pc_w    = (const float*)d_in[9];
  const float* pc_b    = (const float*)d_in[10];
  const float* hd_w    = (const float*)d_in[11];
  const float* hd_b    = (const float*)d_in[12];
  float* out = (float*)d_out;

  // workspace layout: packed pc_w (128 KB) | h0 (512 KB) | c0 (512 KB)
  unsigned* pcwp = (unsigned*)d_ws;
  float* h0 = (float*)((char*)d_ws + 131072);
  float* c0 = h0 + BATCH * NHID;

  pack_pc_kernel<<<128, 256, 0, stream>>>(pc_w, pcwp);
  init_hc_kernel<<<512, 256, 0, stream>>>(init, state_w, state_b, cell_w, cell_b,
                                          h0, c0);

  constexpr size_t SMEM_BYTES =
      (32768 + 16384 + 2048 + 2048 + 4096) * 4 +   // packed weights + act bufs
      (1536 + 512 + 256 + 16 + 96) * 4;            // fp32 wx/biases/x tile
  lstm_kernel<<<32, 256, SMEM_BYTES, stream>>>(x, rnn_w, rnn_b, bneck_w, pc_b,
                                               hd_w, hd_b, pcwp, h0, c0, out);
}
